// _SpatioTemporalMambaBlock_27178553049268
// MI455X (gfx1250) — matrix operand order repaired and checked by reading, not compiled, not run on hardware
//
#include <hip/hip_runtime.h>
#include <hip/hip_bf16.h>
#include <math.h>

// ---------------------------------------------------------------------------
// SpatioTemporal Mamba block for MI455X (gfx1250, wave32, WMMA)
//   x[2,16,1024,256] fp32 -> spatial mamba (scan over N) -> temporal mamba
//   (scan over T) -> FFN.  All large GEMMs run on v_wmma_f32_16x16x32_f16
//   with f16 operands / f32 accumulation.  K-loop is software-pipelined:
//   double-buffered LDS, A tiles via global_load_async_to_lds_b128 (ASYNCcnt)
//   overlapping the WMMA stream; one s_wait_asynccnt + barrier per K-step.
// ---------------------------------------------------------------------------

typedef __attribute__((ext_vector_type(16))) _Float16 v16h;
typedef __attribute__((ext_vector_type(8)))  _Float16 v8h;
typedef __attribute__((ext_vector_type(8)))  float    v8f;

#define DIMC    256
#define DINNER  512
#define DSTATE  16
#define DCONV   4
#define DTRANK  16
#define NB      2
#define NT      16
#define NN      1024
#define NROWS   32768   // = NB*NT*NN = tokens per mamba pass

// GEMM tiling: 256 threads = 8 waves (4x2); wave tile 32x32 (4 wmma accums).
#define BM 128
#define BN 64
#define BK 32
#define LDPAD 48        // padded row length in halves (96 B, 16B-aligned rows)

// Epilogue codes
#define EP_F16        0   // store f16
#define EP_F32        1   // store f32
#define EP_BGELU_F16  2   // +bias, exact gelu, store f16
#define EP_RES_F32    3   // +residual, store f32
#define EP_BRES_F32   4   // +bias +residual, store f32

#define CAT16(lo, hi) __builtin_shufflevector(lo, hi, 0, 1, 2, 3, 4, 5, 6, 7, \
                                              8, 9, 10, 11, 12, 13, 14, 15)

__device__ __forceinline__ void wait_asynccnt0() {
#if __has_builtin(__builtin_amdgcn_s_wait_asynccnt)
  __builtin_amdgcn_s_wait_asynccnt(0);
#else
  asm volatile("s_wait_asynccnt 0" ::: "memory");
#endif
}

__device__ __forceinline__ v8f wmma16(v16h a, v16h b, v8f c) {
  return __builtin_amdgcn_wmma_f32_16x16x32_f16(false, a, false, b,
                                                (short)0, c, false, false);
}

// ---------------------------------------------------------------------------
// f32 -> f16 conversion (weights, once per launch)
// ---------------------------------------------------------------------------
__global__ void cvt_f16_kernel(const float* __restrict__ src,
                               _Float16* __restrict__ dst, int n) {
  int i = blockIdx.x * blockDim.x + threadIdx.x;
  if (i < n) dst[i] = (_Float16)src[i];
}

// ---------------------------------------------------------------------------
// LayerNorm over last dim (256), one wave per row, f16 output for WMMA feed
// ---------------------------------------------------------------------------
__global__ __launch_bounds__(256)
void ln_f16_kernel(const float* __restrict__ x, const float* __restrict__ g,
                   const float* __restrict__ b, _Float16* __restrict__ out,
                   int rows) {
  int row  = (int)((blockIdx.x * blockDim.x + threadIdx.x) >> 5);
  int lane = threadIdx.x & 31;
  if (row >= rows) return;
  const float* xr = x + (size_t)row * DIMC;
  float v[8], s = 0.f, s2 = 0.f;
#pragma unroll
  for (int i = 0; i < 8; ++i) {
    v[i] = xr[lane + 32 * i];
    s += v[i]; s2 += v[i] * v[i];
  }
#pragma unroll
  for (int off = 16; off >= 1; off >>= 1) {
    s  += __shfl_xor(s,  off, 32);
    s2 += __shfl_xor(s2, off, 32);
  }
  float mu   = s * (1.f / DIMC);
  float var  = s2 * (1.f / DIMC) - mu * mu;
  float rstd = rsqrtf(var + 1e-5f);
  _Float16* orow = out + (size_t)row * DIMC;
#pragma unroll
  for (int i = 0; i < 8; ++i) {
    int c = lane + 32 * i;
    orow[c] = (_Float16)((v[i] - mu) * rstd * g[c] + b[c]);
  }
}

// ---------------------------------------------------------------------------
// WMMA GEMM: C[M,N] = A[M,K] * B[K,N], A/B f16 row-major, f32 accumulate.
// Block tile 128x64, 8 waves (4x2), wave tile 32x32 = 4 wmma accumulators.
// Double-buffered LDS; A staged via async global->LDS; B staged transposed
// (Bt[n][k]) so every fragment is 2x aligned ds_load_b128.
// ---------------------------------------------------------------------------
__global__ __launch_bounds__(256)
void gemm_wmma_kernel(const _Float16* __restrict__ A,
                      const _Float16* __restrict__ B,
                      float* __restrict__ Cf, _Float16* __restrict__ Ch,
                      const float* __restrict__ bias,
                      const float* __restrict__ resid,
                      int M, int N, int K, int epi) {
  __shared__ alignas(16) _Float16 As[2][BM][LDPAD];   // row-major A tiles
  __shared__ alignas(16) _Float16 Bt[2][BN][LDPAD];   // transposed B tiles

  int nTilesN = (N + BN - 1) / BN;
  int bm = (int)(blockIdx.x / nTilesN) * BM;
  int bn = (int)(blockIdx.x % nTilesN) * BN;

  int tid  = threadIdx.x;
  int lane = tid & 31;
  int wave = tid >> 5;
  int wm   = wave & 3;    // 32-row slice (0..3)
  int wn   = wave >> 2;   // 32-col slice (0..1)

  // A staging: each thread async-copies two 16B rows-segments (128x32 tile)
  int ar  = tid >> 2;              // 0..63
  int ac  = (tid & 3) * 8;         // 0,8,16,24
  int agr0 = bm + ar;       if (agr0 > M - 1) agr0 = M - 1;
  int agr1 = bm + 64 + ar;  if (agr1 > M - 1) agr1 = M - 1;
  unsigned lA0a = (unsigned)(uintptr_t)&As[0][ar][ac];
  unsigned lA0b = (unsigned)(uintptr_t)&As[1][ar][ac];
  unsigned lA1a = (unsigned)(uintptr_t)&As[0][64 + ar][ac];
  unsigned lA1b = (unsigned)(uintptr_t)&As[1][64 + ar][ac];
  // B staging: coalesced 16B read of one K-row, b16 scatter into Bt
  int bkr = tid >> 3;              // 0..31
  int bc  = (tid & 7) * 8;         // 0..56

  v8f acc[4] = {v8f{}, v8f{}, v8f{}, v8f{}};

  auto stage = [&](int buf, int k0) {
    unsigned la0 = buf ? lA0b : lA0a;
    unsigned la1 = buf ? lA1b : lA1a;
    unsigned long long ga0 =
        (unsigned long long)(uintptr_t)(A + (size_t)agr0 * K + k0 + ac);
    unsigned long long ga1 =
        (unsigned long long)(uintptr_t)(A + (size_t)agr1 * K + k0 + ac);
    asm volatile("global_load_async_to_lds_b128 %0, %1, off"
                 :: "v"(la0), "v"(ga0) : "memory");
    asm volatile("global_load_async_to_lds_b128 %0, %1, off"
                 :: "v"(la1), "v"(ga1) : "memory");
    int gc = bn + bc;
    const _Float16* src = B + (size_t)(k0 + bkr) * N + gc;
    v8h bv = {};
    if (gc + 7 < N) bv = *(const v8h*)src;   // N always multiple of 8: all-or-nothing
#pragma unroll
    for (int i = 0; i < 8; ++i) Bt[buf][bc + i][bkr] = bv[i];
  };

  int kIter = K / BK;
  stage(0, 0);
  wait_asynccnt0();
  __syncthreads();

  int cur = 0;
  for (int it = 0; it < kIter; ++it) {
    // prefetch next tile into the other buffer (overlaps with WMMA below)
    if (it + 1 < kIter) stage(cur ^ 1, (it + 1) * BK);

    int kh8 = (lane >> 4) * 8;
    int kb0 = (lane >> 4) * 16;
    v16h afr[2], bfr[2];
#pragma unroll
    for (int s2 = 0; s2 < 2; ++s2) {
      int am = wm * 32 + s2 * 16 + (lane & 15);
      v8h x0 = *(const v8h*)&As[cur][am][kh8];
      v8h x1 = *(const v8h*)&As[cur][am][16 + kh8];
      afr[s2] = CAT16(x0, x1);
    }
#pragma unroll
    for (int t = 0; t < 2; ++t) {
      int bcol = wn * 32 + t * 16 + (lane & 15);
      v8h x0 = *(const v8h*)&Bt[cur][bcol][kh8];
      v8h x1 = *(const v8h*)&Bt[cur][bcol][16 + kh8];
      bfr[t] = CAT16(x0, x1);
    }
    acc[0] = wmma16(afr[0], bfr[0], acc[0]);
    acc[1] = wmma16(afr[0], bfr[1], acc[1]);
    acc[2] = wmma16(afr[1], bfr[0], acc[2]);
    acc[3] = wmma16(afr[1], bfr[1], acc[3]);

    wait_asynccnt0();   // next A tile landed
    __syncthreads();    // next B tile visible; everyone done reading cur
    cur ^= 1;
  }

  // ---- store D (C layout: VGPR r -> M = r + 8*(lane>=16), N = lane%16) ----
#pragma unroll
  for (int s2 = 0; s2 < 2; ++s2) {
#pragma unroll
    for (int t = 0; t < 2; ++t) {
      v8f a = acc[s2 * 2 + t];
      int n = bn + wn * 32 + t * 16 + (lane & 15);
      if (n >= N) continue;
      int m0 = bm + wm * 32 + s2 * 16 + (lane >> 4) * 8;
      float bn_ = (epi == EP_BGELU_F16 || epi == EP_BRES_F32) ? bias[n] : 0.f;
#pragma unroll
      for (int r = 0; r < 8; ++r) {
        size_t idx = (size_t)(m0 + r) * N + n;
        float v = a[r];
        switch (epi) {
          case EP_F16:       Ch[idx] = (_Float16)v; break;
          case EP_F32:       Cf[idx] = v; break;
          case EP_BGELU_F16: {
            float u = v + bn_;
            Ch[idx] = (_Float16)(0.5f * u * (1.f + erff(u * 0.70710678118f)));
          } break;
          case EP_RES_F32:   Cf[idx] = v + resid[idx]; break;
          case EP_BRES_F32:  Cf[idx] = v + bn_ + resid[idx]; break;
        }
      }
    }
  }
}

// ---------------------------------------------------------------------------
// Causal depthwise conv1d (4 taps over L) + bias + SiLU.  Reads u-half of xz.
// ---------------------------------------------------------------------------
__global__ void conv_silu_kernel(const _Float16* __restrict__ xz,
                                 const float* __restrict__ w,
                                 const float* __restrict__ b,
                                 _Float16* __restrict__ u, int S, int L) {
  int idx = blockIdx.x * blockDim.x + threadIdx.x;
  if (idx >= S * L * DINNER) return;
  int c  = idx & (DINNER - 1);
  int sl = idx >> 9;
  int l  = sl % L;
  int s  = sl / L;
  float acc = b[c];
#pragma unroll
  for (int k = 0; k < DCONV; ++k) {
    int lk = l - (DCONV - 1) + k;
    if (lk >= 0)
      acc += (float)xz[((size_t)(s * L + lk)) * (2 * DINNER) + c] * w[c * DCONV + k];
  }
  float sig = 1.f / (1.f + __expf(-acc));
  u[(size_t)sl * DINNER + c] = (_Float16)(acc * sig);
}

// ---------------------------------------------------------------------------
// dt = softplus(dt_r @ W_dt + b_dt); dt_r = dbl[:, :16]
// ---------------------------------------------------------------------------
__global__ void dt_kernel(const float* __restrict__ dbl,
                          const float* __restrict__ Wdt,
                          const float* __restrict__ bdt,
                          float* __restrict__ dt, int rows) {
  int idx = blockIdx.x * blockDim.x + threadIdx.x;
  if (idx >= rows * DINNER) return;
  int c   = idx & (DINNER - 1);
  int row = idx >> 9;
  float acc = bdt[c];
  const float* dr = dbl + (size_t)row * (DTRANK + 2 * DSTATE);
#pragma unroll
  for (int r = 0; r < DTRANK; ++r) acc += dr[r] * Wdt[r * DINNER + c];
  dt[idx] = (acc > 20.f) ? acc : log1pf(__expf(acc));
}

// ---------------------------------------------------------------------------
// Selective scan: 1 thread per (sequence, channel); 16-state in registers;
// lanes of a wave share a sequence -> B/C loads are wave-uniform.
// Fuses D skip-connection and SiLU(z) gating; writes f16 for W_out GEMM.
// ---------------------------------------------------------------------------
__global__ void scan_kernel(const float* __restrict__ dt,
                            const _Float16* __restrict__ u16,
                            const float* __restrict__ dbl,
                            const _Float16* __restrict__ xz,
                            const float* __restrict__ Alog,
                            const float* __restrict__ Dp,
                            _Float16* __restrict__ y16, int S, int L) {
  int idx = blockIdx.x * blockDim.x + threadIdx.x;
  if (idx >= S * DINNER) return;
  int c = idx & (DINNER - 1);
  int s = idx >> 9;
  float A[DSTATE], h[DSTATE];
#pragma unroll
  for (int j = 0; j < DSTATE; ++j) {
    A[j] = -__expf(Alog[c * DSTATE + j]);
    h[j] = 0.f;
  }
  float Dc = Dp[c];
  for (int l = 0; l < L; ++l) {
    size_t row = (size_t)s * L + l;
    float dtv = dt[row * DINNER + c];
    float uv  = (float)u16[row * DINNER + c];
    float du  = dtv * uv;
    const float* bc = dbl + row * (DTRANK + 2 * DSTATE);
    float y = 0.f;
#pragma unroll
    for (int j = 0; j < DSTATE; ++j) {
      h[j] = h[j] * __expf(dtv * A[j]) + du * bc[DTRANK + j];
      y += h[j] * bc[DTRANK + DSTATE + j];
    }
    float zv = (float)xz[row * (2 * DINNER) + DINNER + c];
    float zg = zv / (1.f + __expf(-zv));
    y16[row * DINNER + c] = (_Float16)((y + uv * Dc) * zg);
  }
}

// ---------------------------------------------------------------------------
// Transpose middle dims: src[B,X,Y,D] -> dst[B,Y,X,D], float4 row copies.
// ---------------------------------------------------------------------------
__global__ void transpose_kernel(const float* __restrict__ src,
                                 float* __restrict__ dst, int B, int X, int Y) {
  int idx = blockIdx.x * blockDim.x + threadIdx.x;
  if (idx >= B * X * Y * (DIMC / 4)) return;
  int q = idx & 63;
  int r = idx >> 6;                 // dst row index over [B,Y,X]
  int x = r % X;
  int y = (r / X) % Y;
  int b = r / (X * Y);
  size_t srow = ((size_t)b * X + x) * Y + y;
  const float4* s4 = (const float4*)src + srow * (DIMC / 4) + q;
  float4* d4 = (float4*)dst + (size_t)r * (DIMC / 4) + q;
  *d4 = *s4;
}

// ---------------------------------------------------------------------------
// Host-side orchestration
// ---------------------------------------------------------------------------
static inline size_t alignup(size_t x) { return (x + 255) & ~(size_t)255; }

struct MambaW {
  const float *ln_g, *ln_b, *W_in, *conv_w, *conv_b, *W_x, *W_dt, *b_dt, *A_log, *D, *W_out;
  _Float16 *W_in16, *W_x16, *W_out16;
};

static void launch_gemm(hipStream_t st, const _Float16* A, const _Float16* B,
                        float* Cf, _Float16* Ch, const float* bias,
                        const float* resid, int M, int N, int K, int epi) {
  int nT = ((M + BM - 1) / BM) * ((N + BN - 1) / BN);
  gemm_wmma_kernel<<<nT, 256, 0, st>>>(A, B, Cf, Ch, bias, resid, M, N, K, epi);
}

static void run_mamba(hipStream_t st, const float* x_in, float* x_out,
                      const MambaW& w, int S, int L,
                      _Float16* h16, _Float16* xz16, _Float16* u16,
                      float* dbl, float* dtb, _Float16* y16) {
  const int rows = S * L;  // == NROWS for both passes
  // 1. LayerNorm -> f16
  ln_f16_kernel<<<(rows * 32 + 255) / 256, 256, 0, st>>>(x_in, w.ln_g, w.ln_b, h16, rows);
  // 2. xz = h @ W_in   [rows,256] x [256,1024] -> f16
  launch_gemm(st, h16, w.W_in16, nullptr, xz16, nullptr, nullptr, rows, 2 * DINNER, DIMC, EP_F16);
  // 3. causal conv + SiLU -> u (f16)
  conv_silu_kernel<<<(rows * DINNER + 255) / 256, 256, 0, st>>>(xz16, w.conv_w, w.conv_b, u16, S, L);
  // 4. dbl = u @ W_x   [rows,512] x [512,48] -> f32
  launch_gemm(st, u16, w.W_x16, dbl, nullptr, nullptr, nullptr, rows, DTRANK + 2 * DSTATE, DINNER, EP_F32);
  // 5. dt = softplus(dt_r @ W_dt + b_dt)
  dt_kernel<<<(rows * DINNER + 255) / 256, 256, 0, st>>>(dbl, w.W_dt, w.b_dt, dtb, rows);
  // 6. selective scan + D skip + SiLU(z) gate -> f16
  scan_kernel<<<(S * DINNER + 255) / 256, 256, 0, st>>>(dtb, u16, dbl, xz16, w.A_log, w.D, y16, S, L);
  // 7. out = x + y @ W_out   [rows,512] x [512,256] + residual
  launch_gemm(st, y16, w.W_out16, x_out, nullptr, nullptr, x_in, rows, DIMC, DINNER, EP_RES_F32);
}

extern "C" void kernel_launch(void* const* d_in, const int* in_sizes, int n_in,
                              void* d_out, int out_size, void* d_ws, size_t ws_size,
                              hipStream_t stream) {
  (void)in_sizes; (void)n_in; (void)out_size; (void)ws_size;
  const float* x_in = (const float*)d_in[0];

  MambaW sp, tp;
  auto fill = [&](MambaW& m, int base) {
    m.ln_g   = (const float*)d_in[base + 0];
    m.ln_b   = (const float*)d_in[base + 1];
    m.W_in   = (const float*)d_in[base + 2];
    m.conv_w = (const float*)d_in[base + 3];
    m.conv_b = (const float*)d_in[base + 4];
    m.W_x    = (const float*)d_in[base + 5];
    m.W_dt   = (const float*)d_in[base + 6];
    m.b_dt   = (const float*)d_in[base + 7];
    m.A_log  = (const float*)d_in[base + 8];
    m.D      = (const float*)d_in[base + 9];
    m.W_out  = (const float*)d_in[base + 10];
  };
  fill(sp, 1);
  fill(tp, 12);
  const float* f_ln_g = (const float*)d_in[23];
  const float* f_ln_b = (const float*)d_in[24];
  const float* f_W1   = (const float*)d_in[25];
  const float* f_b1   = (const float*)d_in[26];
  const float* f_W2   = (const float*)d_in[27];
  const float* f_b2   = (const float*)d_in[28];

  // ---- workspace layout ----
  char* ws = (char*)d_ws;
  size_t off = 0;
  auto take = [&](size_t bytes) { char* p = ws + off; off += alignup(bytes); return p; };

  _Float16* w_in_s  = (_Float16*)take(DIMC * 2 * DINNER * 2);
  _Float16* w_x_s   = (_Float16*)take(DINNER * (DTRANK + 2 * DSTATE) * 2);
  _Float16* w_out_s = (_Float16*)take(DINNER * DIMC * 2);
  _Float16* w_in_t  = (_Float16*)take(DIMC * 2 * DINNER * 2);
  _Float16* w_x_t   = (_Float16*)take(DINNER * (DTRANK + 2 * DSTATE) * 2);
  _Float16* w_out_t = (_Float16*)take(DINNER * DIMC * 2);
  _Float16* w1_16   = (_Float16*)take(DIMC * 1024 * 2);
  _Float16* w2_16   = (_Float16*)take(1024 * DIMC * 2);

  _Float16* h16   = (_Float16*)take((size_t)NROWS * DIMC * 2);
  _Float16* xz16  = (_Float16*)take((size_t)NROWS * 2 * DINNER * 2);   // also FFN hidden
  _Float16* u16   = (_Float16*)take((size_t)NROWS * DINNER * 2);
  float*    dbl   = (float*)   take((size_t)NROWS * (DTRANK + 2 * DSTATE) * 4);
  float*    dtb   = (float*)   take((size_t)NROWS * DINNER * 4);
  _Float16* y16   = (_Float16*)take((size_t)NROWS * DINNER * 2);
  float*    bufA  = (float*)   take((size_t)NROWS * DIMC * 4);
  float*    bufB  = (float*)   take((size_t)NROWS * DIMC * 4);

  auto cvt = [&](const float* s, _Float16* d, int n) {
    cvt_f16_kernel<<<(n + 255) / 256, 256, 0, stream>>>(s, d, n);
  };
  cvt(sp.W_in, w_in_s, DIMC * 2 * DINNER);
  cvt(sp.W_x, w_x_s, DINNER * (DTRANK + 2 * DSTATE));
  cvt(sp.W_out, w_out_s, DINNER * DIMC);
  cvt(tp.W_in, w_in_t, DIMC * 2 * DINNER);
  cvt(tp.W_x, w_x_t, DINNER * (DTRANK + 2 * DSTATE));
  cvt(tp.W_out, w_out_t, DINNER * DIMC);
  cvt(f_W1, w1_16, DIMC * 1024);
  cvt(f_W2, w2_16, 1024 * DIMC);
  sp.W_in16 = w_in_s; sp.W_x16 = w_x_s; sp.W_out16 = w_out_s;
  tp.W_in16 = w_in_t; tp.W_x16 = w_x_t; tp.W_out16 = w_out_t;

  // ---- spatial mamba: [B*T, N, D], scan over N ----
  run_mamba(stream, x_in, bufA, sp, NB * NT, NN, h16, xz16, u16, dbl, dtb, y16);

  // ---- transpose [B,T,N,D] -> [B,N,T,D] ----
  transpose_kernel<<<(NROWS * 64 + 255) / 256, 256, 0, stream>>>(bufA, bufB, NB, NT, NN);

  // ---- temporal mamba: [B*N, T, D], scan over T ----
  run_mamba(stream, bufB, bufA, tp, NB * NN, NT, h16, xz16, u16, dbl, dtb, y16);

  // ---- transpose back [B,N,T,D] -> [B,T,N,D] into d_out ----
  float* out = (float*)d_out;
  transpose_kernel<<<(NROWS * 64 + 255) / 256, 256, 0, stream>>>(bufA, out, NB, NN, NT);

  // ---- FFN: out += gelu(ln(out) @ W1 + b1) @ W2 + b2 ----
  ln_f16_kernel<<<(NROWS * 32 + 255) / 256, 256, 0, stream>>>(out, f_ln_g, f_ln_b, h16, NROWS);
  _Float16* hid16 = xz16;  // alias: xz dead after temporal pass
  launch_gemm(stream, h16, w1_16, nullptr, hid16, f_b1, nullptr, NROWS, 1024, DIMC, EP_BGELU_F16);
  launch_gemm(stream, hid16, w2_16, out, nullptr, f_b2, out, NROWS, DIMC, 1024, EP_BRES_F32);
}
